// ClusterGNN_21973052686418
// MI455X (gfx1250) — compile-verified, matrix operand order checked by reading, not run on hardware
//
#include <hip/hip_runtime.h>
#include <hip/hip_bf16.h>

// ---------------------------------------------------------------------------
// Problem constants (match reference)
// ---------------------------------------------------------------------------
static constexpr int N_NODES = 100000;
static constexpr int N_EDGES = 640000;
static constexpr int DH      = 128;       // D == H == 128
static constexpr int BATCH   = 16;
static constexpr int C0      = 1600;      // B * 100
static constexpr int C1      = 800;       // B * 50
static constexpr int C2      = 160;       // B * 10

typedef __attribute__((ext_vector_type(16))) __bf16 v16bf;
typedef __attribute__((ext_vector_type(8)))  float  v8f;

__device__ __forceinline__ __bf16 f2bf(float f) { return (__bf16)f; }

// K-index interleave for 16-bit A/B fragments (ISA 7.12.2, 16-bit A 16x32):
//   g = lane/16, v = vgpr index 0..7, h = half 0..1
//   K = (v&3)*2 + h + g*8 + (v>>2)*16
__device__ __forceinline__ int kbase(int v, int g) {
    return ((v & 3) << 1) + (g << 3) + ((v >> 2) << 4);
}

// ---------------------------------------------------------------------------
// Pack one 128x128 fp32 weight (row-major, K x N) into WMMA B-fragment layout:
// Wp[(ks*8 + ntile)*32 + lane] = v16bf (16 bf16 = 32 bytes per lane).
// Lane holds column N = ntile*16 + lane%16, K per kbase() map.
// ---------------------------------------------------------------------------
__global__ void pack_w_frags(const float* __restrict__ W, v16bf* __restrict__ Wp) {
    int idx = blockIdx.x * blockDim.x + threadIdx.x;
    if (idx >= 32 * 32) return;
    int frag = idx >> 5;      // ks*8 + ntile
    int lane = idx & 31;
    int ks = frag >> 3, nt = frag & 7;
    int g = lane >> 4;
    int col = nt * 16 + (lane & 15);
    v16bf b;
#pragma unroll
    for (int v = 0; v < 8; ++v) {
        int kb = ks * 32 + kbase(v, g);
        b[2 * v]     = f2bf(W[(size_t)kb * DH + col]);
        b[2 * v + 1] = f2bf(W[(size_t)(kb + 1) * DH + col]);
    }
    Wp[frag * 32 + lane] = b;
}

// ---------------------------------------------------------------------------
// Y[M x 128] = X[M x 128] @ W  (W pre-packed bf16 fragments; f32 accumulate)
// Packed W is staged once per workgroup into LDS (32 KB); 8 waves share it.
// One wave -> one 16-row x 128-col output tile. 32x v_wmma_f32_16x16x32_bf16.
// M must be a multiple of 16 (100000, 1600, 800, 160 all are).
// ---------------------------------------------------------------------------
__global__ __launch_bounds__(256) void gemm_xw_wmma(
    const float* __restrict__ X, const v16bf* __restrict__ Wp,
    float* __restrict__ Y, int Mtiles) {
    __shared__ v16bf sW[32 * 32];                    // 32 KB
    for (int i = threadIdx.x; i < 32 * 32; i += 256) sW[i] = Wp[i];
    __syncthreads();

    int wave = threadIdx.x >> 5;
    int lane = threadIdx.x & 31;
    int tile = blockIdx.x * (blockDim.x >> 5) + wave;
    if (tile >= Mtiles) return;          // wave-uniform: EXEC stays all-ones

    int g = lane >> 4;
    int mrow = tile * 16 + (lane & 15);
    const float* xr = X + (size_t)mrow * DH;

    v8f acc[8] = {};
#pragma unroll
    for (int ks = 0; ks < 4; ++ks) {
        v16bf a;
#pragma unroll
        for (int v = 0; v < 8; ++v) {
            int kb = ks * 32 + kbase(v, g);
            a[2 * v]     = f2bf(xr[kb]);
            a[2 * v + 1] = f2bf(xr[kb + 1]);
        }
#pragma unroll
        for (int nt = 0; nt < 8; ++nt) {
            v16bf b = sW[(ks * 8 + nt) * 32 + lane];   // ds_load_b128 x2
            acc[nt] = __builtin_amdgcn_wmma_f32_16x16x32_bf16(
                false, a, false, b, (short)0, acc[nt], false, false);
        }
    }
    // C/D layout: VGPR r holds M = r + 8*g, N = lane%16 (+ ntile*16)
#pragma unroll
    for (int nt = 0; nt < 8; ++nt) {
        float* yr = Y + (size_t)(tile * 16 + 8 * g) * DH + nt * 16 + (lane & 15);
#pragma unroll
        for (int r = 0; r < 8; ++r) yr[(size_t)r * DH] = acc[nt][r];
    }
}

// ---------------------------------------------------------------------------
// Sparse-GCN helper kernels (layer 0)
// ---------------------------------------------------------------------------
__global__ void init_deg(float* deg, int n) {
    int i = blockIdx.x * blockDim.x + threadIdx.x;
    if (i < n) deg[i] = 1.0f;
}
__global__ void count_deg(const int* __restrict__ dst, float* deg, int e) {
    int i = blockIdx.x * blockDim.x + threadIdx.x;
    if (i < e) atomicAdd(&deg[dst[i]], 1.0f);
}
__global__ void calc_dis(const float* __restrict__ deg, float* dis, int n) {
    int i = blockIdx.x * blockDim.x + threadIdx.x;
    if (i < n) dis[i] = rsqrtf(deg[i]);
}
__global__ void edge_coeff(const int* __restrict__ src, const int* __restrict__ dst,
                           const float* __restrict__ dis, float* cf, int e) {
    int i = blockIdx.x * blockDim.x + threadIdx.x;
    if (i < e) cf[i] = dis[src[i]] * dis[dst[i]];
}
// one thread = one (edge, 4-dim group): float4 gather + 4 L2-resident atomics
__global__ void scatter_msg4(const int* __restrict__ src, const int* __restrict__ dst,
                             const float* __restrict__ cf, const float* __restrict__ h,
                             float* agg, long long total) {
    long long i = (long long)blockIdx.x * blockDim.x + threadIdx.x;
    if (i >= total) return;                   // total = E * 32
    int e = (int)(i >> 5), q = (int)(i & 31);
    float  c  = cf[e];
    float4 hv = ((const float4*)(h + (size_t)src[e] * DH))[q];
    float* ap = agg + (size_t)dst[e] * DH + q * 4;
    atomicAdd(ap + 0, hv.x * c);
    atomicAdd(ap + 1, hv.y * c);
    atomicAdd(ap + 2, hv.z * c);
    atomicAdd(ap + 3, hv.w * c);
}
// agg <- relu(agg + h/deg + b); fused cluster-mean-pool accumulation (float4)
__global__ void finalize0_pool4(float* __restrict__ agg, const float* __restrict__ h,
                                const float* __restrict__ deg, const float* __restrict__ bias,
                                const int* __restrict__ lab, float* xs, float* cnt,
                                long long total) {
    long long i = (long long)blockIdx.x * blockDim.x + threadIdx.x;
    if (i >= total) return;                   // total = N * 32
    int node = (int)(i >> 5), q = (int)(i & 31);
    float  dg = deg[node];
    float4 av = ((const float4*)agg)[i];
    float4 hv = ((const float4*)h)[i];
    float4 bv = ((const float4*)bias)[q];
    float4 v;
    v.x = fmaxf(av.x + hv.x / dg + bv.x, 0.0f);
    v.y = fmaxf(av.y + hv.y / dg + bv.y, 0.0f);
    v.z = fmaxf(av.z + hv.z / dg + bv.z, 0.0f);
    v.w = fmaxf(av.w + hv.w / dg + bv.w, 0.0f);
    ((float4*)agg)[i] = v;
    int c = lab[node];
    float* xp = xs + (size_t)c * DH + q * 4;
    atomicAdd(xp + 0, v.x);
    atomicAdd(xp + 1, v.y);
    atomicAdd(xp + 2, v.z);
    atomicAdd(xp + 3, v.w);
    if (q == 0) atomicAdd(&cnt[c], 1.0f);
}
__global__ void pool_div(float* xs, const float* __restrict__ cnt, int total) {
    int i = blockIdx.x * blockDim.x + threadIdx.x;
    if (i >= total) return;
    float c = cnt[i >> 7];
    xs[i] = (c > 0.0f) ? xs[i] / c : 0.0f;
}
// adjacency stored TRANSPOSED: AT[dstLabel * C + srcLabel] = 1
__global__ void build_A1T(const int* __restrict__ src, const int* __restrict__ dst,
                          const int* __restrict__ lab, float* AT, int e) {
    int i = blockIdx.x * blockDim.x + threadIdx.x;
    if (i >= e) return;
    int s = lab[src[i]], d = lab[dst[i]];
    if (s != d) AT[(size_t)d * C0 + s] = 1.0f;   // races benign (all write 1)
}

// ---------------------------------------------------------------------------
// Dense-GCN helper kernels (adjacency kept transposed: AT[c, r] = A[r, c])
// ---------------------------------------------------------------------------
// dis[c] = rsqrt(1 + sum_r A[r,c]) = rsqrt(1 + sum_r AT[c,r]); one block per c,
// contiguous coalesced row scan + LDS tree reduction.
__global__ void colsum_dis_T(const float* __restrict__ AT, float* dis, int C) {
    __shared__ float red[128];
    int c = blockIdx.x;
    float s = 0.0f;
    for (int r = threadIdx.x; r < C; r += blockDim.x) s += AT[(size_t)c * C + r];
    red[threadIdx.x] = s;
    __syncthreads();
    for (int k = 64; k > 0; k >>= 1) {
        if (threadIdx.x < k) red[threadIdx.x] += red[threadIdx.x + k];
        __syncthreads();
    }
    if (threadIdx.x == 0) dis[c] = rsqrtf(1.0f + red[0]);
}
__global__ void scale_rows(float* h, const float* __restrict__ dis, int total) {
    int i = blockIdx.x * blockDim.x + threadIdx.x;
    if (i < total) h[i] *= dis[i >> 7];
}
// Y[c,:] = ( Z[c,:] + sum_{r: AT[c,r]!=0} Z[r,:] ) * dis[c] + b ; optional relu
// One 128-thread block per column; AT row is streamed contiguously and is
// block-uniform -> scalar loads + wave-uniform skip branch.
__global__ void dense_agg(const float* __restrict__ AT, const float* __restrict__ Z,
                          const float* __restrict__ dis, const float* __restrict__ bias,
                          float* __restrict__ Y, int C, int relu) {
    int c = blockIdx.x, d = threadIdx.x;
    const float* arow = AT + (size_t)c * C;
    float acc = Z[(size_t)c * DH + d];       // identity (self loop) term
    for (int r = 0; r < C; ++r) {
        if (arow[r] != 0.0f) acc += Z[(size_t)r * DH + d];
    }
    float y = acc * dis[c] + bias[d];
    if (relu) y = fmaxf(y, 0.0f);
    Y[(size_t)c * DH + d] = y;
}
__global__ void pool_sum4(const float* __restrict__ Y, const int* __restrict__ lab,
                          float* xs, float* cnt, int total) {
    int i = blockIdx.x * blockDim.x + threadIdx.x;
    if (i >= total) return;                   // total = C * 32
    int node = i >> 5, q = i & 31;
    int c = lab[node];
    float4 v = ((const float4*)Y)[i];
    float* xp = xs + (size_t)c * DH + q * 4;
    atomicAdd(xp + 0, v.x);
    atomicAdd(xp + 1, v.y);
    atomicAdd(xp + 2, v.z);
    atomicAdd(xp + 3, v.w);
    if (q == 0) atomicAdd(&cnt[c], 1.0f);
}
// Coarsen: orientation-invariant under the transposed convention:
// AT2[l(r), l(c)] = 1 wherever AT1[r, c] != 0 and labels differ.
__global__ void coarsen_k(const float* __restrict__ AT, const int* __restrict__ lab,
                          float* ATn, int C, int Cn) {
    long long i = (long long)blockIdx.x * blockDim.x + threadIdx.x;
    if (i >= (long long)C * C) return;
    if (AT[i] != 0.0f) {
        int r = (int)(i / C), c = (int)(i % C);
        int a = lab[r], b = lab[c];
        if (a != b) ATn[(size_t)a * Cn + b] = 1.0f;
    }
}
__global__ void graph_pool(const float* __restrict__ Y3, float* out) {
    int i = blockIdx.x * blockDim.x + threadIdx.x;
    if (i >= BATCH * DH) return;
    int b = i >> 7, d = i & 127;
    float s = 0.0f;
    for (int k = 0; k < 10; ++k) s += Y3[(size_t)(b * 10 + k) * DH + d];
    out[i] = s * 0.1f;
}

// ---------------------------------------------------------------------------
// Host orchestration
// ---------------------------------------------------------------------------
extern "C" void kernel_launch(void* const* d_in, const int* in_sizes, int n_in,
                              void* d_out, int out_size, void* d_ws, size_t ws_size,
                              hipStream_t stream) {
    const float* x   = (const float*)d_in[0];
    const int*   ei  = (const int*)d_in[1];
    const int*   src = ei;
    const int*   dst = ei + N_EDGES;
    const int*   lab0 = (const int*)d_in[3];
    const int*   lab1 = (const int*)d_in[4];
    const int*   lab2 = (const int*)d_in[5];
    const float* W[4] = { (const float*)d_in[6], (const float*)d_in[8],
                          (const float*)d_in[10], (const float*)d_in[12] };
    const float* b[4] = { (const float*)d_in[7], (const float*)d_in[9],
                          (const float*)d_in[11], (const float*)d_in[13] };
    float* out = (float*)d_out;

    // --- workspace layout ---------------------------------------------------
    char* base = (char*)d_ws;
    size_t off = 0;
    auto alloc = [&](size_t bytes) -> char* {
        char* p = base + off;
        off = (off + bytes + 255) & ~(size_t)255;
        return p;
    };
    float*  h    = (float*)alloc((size_t)N_NODES * DH * 4);
    float*  agg  = (float*)alloc((size_t)N_NODES * DH * 4);   // -> becomes h0
    float*  deg  = (float*)alloc((size_t)N_NODES * 4);
    float*  dis  = (float*)alloc((size_t)N_NODES * 4);
    float*  cf   = (float*)alloc((size_t)N_EDGES * 4);
    v16bf*  Wp[4];
    for (int i = 0; i < 4; ++i) Wp[i] = (v16bf*)alloc(32 * 32 * sizeof(v16bf));
    // zero-initialized region (contiguous): pooled sums/counts + adjacencies
    char*  zbeg = base + off;
    float* x1s  = (float*)alloc((size_t)C0 * DH * 4);
    float* cnt0 = (float*)alloc((size_t)C0 * 4);
    float* A1   = (float*)alloc((size_t)C0 * C0 * 4);   // transposed convention
    float* x2s  = (float*)alloc((size_t)C1 * DH * 4);
    float* cnt1 = (float*)alloc((size_t)C1 * 4);
    float* A2   = (float*)alloc((size_t)C1 * C1 * 4);
    float* x3s  = (float*)alloc((size_t)C2 * DH * 4);
    float* cnt2 = (float*)alloc((size_t)C2 * 4);
    float* A3   = (float*)alloc((size_t)C2 * C2 * 4);
    char*  zend = base + off;
    float* h1   = (float*)alloc((size_t)C0 * DH * 4);
    float* y1   = (float*)alloc((size_t)C0 * DH * 4);
    float* dis1 = (float*)alloc((size_t)C0 * 4);
    float* h2   = (float*)alloc((size_t)C1 * DH * 4);
    float* y2   = (float*)alloc((size_t)C1 * DH * 4);
    float* dis2 = (float*)alloc((size_t)C1 * 4);
    float* h3   = (float*)alloc((size_t)C2 * DH * 4);
    float* y3   = (float*)alloc((size_t)C2 * DH * 4);
    float* dis3 = (float*)alloc((size_t)C2 * 4);
    if (off > ws_size) return;   // not enough scratch; nothing safe to do

    const int T = 256;
    auto blocks = [](long long n, int t) { return (unsigned)((n + t - 1) / t); };

    // zero accumulators / adjacencies (runs every call -> deterministic)
    hipMemsetAsync(agg, 0, (size_t)N_NODES * DH * 4, stream);
    hipMemsetAsync(zbeg, 0, (size_t)(zend - zbeg), stream);

    // pack weights into WMMA fragment layout
    for (int i = 0; i < 4; ++i)
        pack_w_frags<<<4, T, 0, stream>>>(W[i], Wp[i]);

    // ---- layer 0: sparse GCN on full graph --------------------------------
    gemm_xw_wmma<<<blocks(N_NODES / 16, 8), T, 0, stream>>>(x, Wp[0], h, N_NODES / 16);
    init_deg<<<blocks(N_NODES, T), T, 0, stream>>>(deg, N_NODES);
    count_deg<<<blocks(N_EDGES, T), T, 0, stream>>>(dst, deg, N_EDGES);
    calc_dis<<<blocks(N_NODES, T), T, 0, stream>>>(deg, dis, N_NODES);
    edge_coeff<<<blocks(N_EDGES, T), T, 0, stream>>>(src, dst, dis, cf, N_EDGES);
    scatter_msg4<<<blocks((long long)N_EDGES * 32, T), T, 0, stream>>>(
        src, dst, cf, h, agg, (long long)N_EDGES * 32);
    finalize0_pool4<<<blocks((long long)N_NODES * 32, T), T, 0, stream>>>(
        agg, h, deg, b[0], lab0, x1s, cnt0, (long long)N_NODES * 32);
    pool_div<<<blocks(C0 * DH, T), T, 0, stream>>>(x1s, cnt0, C0 * DH);
    build_A1T<<<blocks(N_EDGES, T), T, 0, stream>>>(src, dst, lab0, A1, N_EDGES);

    // ---- layer 1: dense GCN on C0 -----------------------------------------
    colsum_dis_T<<<C0, 128, 0, stream>>>(A1, dis1, C0);
    gemm_xw_wmma<<<blocks(C0 / 16, 8), T, 0, stream>>>(x1s, Wp[1], h1, C0 / 16);
    scale_rows<<<blocks(C0 * DH, T), T, 0, stream>>>(h1, dis1, C0 * DH);
    dense_agg<<<C0, DH, 0, stream>>>(A1, h1, dis1, b[1], y1, C0, 1);
    pool_sum4<<<blocks(C0 * 32, T), T, 0, stream>>>(y1, lab1, x2s, cnt1, C0 * 32);
    pool_div<<<blocks(C1 * DH, T), T, 0, stream>>>(x2s, cnt1, C1 * DH);
    coarsen_k<<<blocks((long long)C0 * C0, T), T, 0, stream>>>(A1, lab1, A2, C0, C1);

    // ---- layer 2: dense GCN on C1 -----------------------------------------
    colsum_dis_T<<<C1, 128, 0, stream>>>(A2, dis2, C1);
    gemm_xw_wmma<<<blocks(C1 / 16, 8), T, 0, stream>>>(x2s, Wp[2], h2, C1 / 16);
    scale_rows<<<blocks(C1 * DH, T), T, 0, stream>>>(h2, dis2, C1 * DH);
    dense_agg<<<C1, DH, 0, stream>>>(A2, h2, dis2, b[2], y2, C1, 1);
    pool_sum4<<<blocks(C1 * 32, T), T, 0, stream>>>(y2, lab2, x3s, cnt2, C1 * 32);
    pool_div<<<blocks(C2 * DH, T), T, 0, stream>>>(x3s, cnt2, C2 * DH);
    coarsen_k<<<blocks((long long)C1 * C1, T), T, 0, stream>>>(A2, lab2, A3, C1, C2);

    // ---- final conv on C2 + per-graph mean --------------------------------
    colsum_dis_T<<<C2, 128, 0, stream>>>(A3, dis3, C2);
    gemm_xw_wmma<<<blocks(C2 / 16, 8), T, 0, stream>>>(x3s, Wp[3], h3, C2 / 16);
    scale_rows<<<blocks(C2 * DH, T), T, 0, stream>>>(h3, dis3, C2 * DH);
    dense_agg<<<C2, DH, 0, stream>>>(A3, h3, dis3, b[3], y3, C2, 0);
    graph_pool<<<blocks(BATCH * DH, T), T, 0, stream>>>(y3, out);
}
